// RNN_Decoder_29472065585936
// MI455X (gfx1250) — matrix operand order repaired and checked by reading, not compiled, not run on hardware
//
#include <hip/hip_runtime.h>
#include <hip/hip_bf16.h>
#include <math.h>

// ---------------------------------------------------------------------------
// MI455X / gfx1250 implementation. All GEMMs use v_wmma_f32_16x16x32_f16
// (fp32 inputs converted to f16 in-register; data is 0.02-scaled so f16 is
// numerically fine; accumulation is f32). wave32.
// Each wave computes a 32x64 C tile: 2 (M) x 4 (N) 16x16 sub-tiles,
// 8 WMMAs per 32-wide K step with only 2 A-frag + 4 B-frag loads.
// ---------------------------------------------------------------------------

typedef __attribute__((ext_vector_type(16))) _Float16 v16h;
typedef __attribute__((ext_vector_type(8)))  float    v8f;

#define BQ 256
#define LQ 64
#define FEATQ 2048
#define UNITSQ 512
#define EMBQ 256
#define VOCABQ 32000

// ---------------------------------------------------------------------------
// Fused GEMM: C[M,N] = act( A[M,K] @ B[K,N] + bias[N] + addrow[row/rowdiv][N] )
// act: 0 = identity, 1 = tanh.
// Wave tile 32x64. 8 waves/block cover 512 N-columns per block row.
// M%32==0, N%64==0 per wave chunk, K%32==0, K%8==0 (float4 A loads).
// ---------------------------------------------------------------------------
__global__ __launch_bounds__(256) void gemm_wmma_f16(
    const float* __restrict__ A, const float* __restrict__ B,
    const float* __restrict__ bias, const float* __restrict__ addrow,
    float* __restrict__ C, int M, int N, int K, int rowdiv, int act)
{
    const int wave  = threadIdx.x >> 5;
    const int lane  = threadIdx.x & 31;
    const int row16 = lane & 15;   // M-sub-row (A) / N-sub-col (B,C)
    const int half  = lane >> 4;   // lane-half: splits K (A,B) and M (C)

    const int m0 = blockIdx.x * 32;
    const int n0 = (blockIdx.y * 8 + wave) * 64;
    if (m0 >= M || n0 >= N) return;

    const float* __restrict__ Arow0 = A + (size_t)(m0 + row16)      * K;
    const float* __restrict__ Arow1 = A + (size_t)(m0 + 16 + row16) * K;
    const float* __restrict__ Bbase = B + (size_t)n0 + row16;

    v8f acc[2][4];
    #pragma unroll
    for (int mi = 0; mi < 2; ++mi)
        #pragma unroll
        for (int ni = 0; ni < 4; ++ni)
            acc[mi][ni] = v8f{0.f,0.f,0.f,0.f,0.f,0.f,0.f,0.f};

    for (int k0 = 0; k0 < K; k0 += 32) {
        if (k0 + 32 < K) {
            __builtin_prefetch(Arow0 + k0 + 32, 0, 0);                // global_prefetch_b8
            __builtin_prefetch(Arow1 + k0 + 32, 0, 0);
            __builtin_prefetch(Bbase + (size_t)(k0 + 32) * N, 0, 0);
        }
        // A fragments: contiguous k order, matching B. Lane's 8 K-values per group g are
        // contiguous at k0 + half*16 + g*8  ->  two float4 loads per group.
        v16h a[2];
        #pragma unroll
        for (int mi = 0; mi < 2; ++mi) {
            const float* __restrict__ Ar = mi ? Arow1 : Arow0;
            #pragma unroll
            for (int g = 0; g < 2; ++g) {
                const float4 p0 = *(const float4*)(Ar + k0 + (half << 4) + g * 8);
                const float4 p1 = *(const float4*)(Ar + k0 + (half << 4) + g * 8 + 4);
                a[mi][g*8+0] = (_Float16)p0.x; a[mi][g*8+1] = (_Float16)p0.y;
                a[mi][g*8+2] = (_Float16)p0.z; a[mi][g*8+3] = (_Float16)p0.w;
                a[mi][g*8+4] = (_Float16)p1.x; a[mi][g*8+5] = (_Float16)p1.y;
                a[mi][g*8+6] = (_Float16)p1.z; a[mi][g*8+7] = (_Float16)p1.w;
            }
        }
        // B fragments per 16-wide N sub-tile; lane holds one column, VGPR v
        // holds K-pair k = half*16 + 2v (rows -> coalesced across lanes).
        #pragma unroll
        for (int ni = 0; ni < 4; ++ni) {
            const float* __restrict__ Bc = Bbase + ni * 16;
            v16h b;
            #pragma unroll
            for (int v = 0; v < 8; ++v) {
                const int k = (half << 4) + (v << 1);
                b[2*v]   = (_Float16)Bc[(size_t)(k0 + k)     * N];
                b[2*v+1] = (_Float16)Bc[(size_t)(k0 + k + 1) * N];
            }
            acc[0][ni] = __builtin_amdgcn_wmma_f32_16x16x32_f16(
                false, a[0], false, b, (short)0, acc[0][ni], false, false);
            acc[1][ni] = __builtin_amdgcn_wmma_f32_16x16x32_f16(
                false, a[1], false, b, (short)0, acc[1][ni], false, false);
        }
    }

    // C layout: VGPR i -> row mbase + half*8 + i, col nbase + row16
    #pragma unroll
    for (int ni = 0; ni < 4; ++ni) {
        const int col = n0 + ni * 16 + row16;
        const float bv = bias ? bias[col] : 0.f;
        #pragma unroll
        for (int mi = 0; mi < 2; ++mi) {
            #pragma unroll
            for (int i = 0; i < 8; ++i) {
                const int r = m0 + mi * 16 + (half << 3) + i;
                float v = acc[mi][ni][i] + bv;
                if (addrow) v += addrow[(size_t)(r / rowdiv) * N + col];
                if (act == 1) v = tanhf(v);
                C[(size_t)r * N + col] = v;
            }
        }
    }
}

// ---------------------------------------------------------------------------
// Attention: per batch b — logits = score[b] @ V + bV ; softmax over L ;
// context[b,f] = sum_l attn[l] * features[b,l,f].  One 256-thread block per b.
// ---------------------------------------------------------------------------
__global__ __launch_bounds__(256) void attention_kernel(
    const float* __restrict__ score,    // (B, L, UNITS)
    const float* __restrict__ features, // (B, L, FEAT)
    const float* __restrict__ Vvec,     // (UNITS)
    const float* __restrict__ bV,       // (1)
    float* __restrict__ attn_out,       // (B, L)
    float* __restrict__ context)        // (B, FEAT)
{
    __shared__ float s_part[LQ][4];
    __shared__ float s_logit[LQ];
    __shared__ float s_attn[LQ];

    const int b = blockIdx.x;
    const int t = threadIdx.x;

    // partial dot products: 4 threads per l, each over 128 elements
    {
        const int l = t >> 2, part = t & 3;
        const float* srow = score + ((size_t)b * LQ + l) * UNITSQ + part * 128;
        const float* vrow = Vvec + part * 128;
        float s = 0.f;
        #pragma unroll 4
        for (int k = 0; k < 128; ++k) s += srow[k] * vrow[k];
        s_part[l][part] = s;
    }
    __syncthreads();
    if (t < LQ)
        s_logit[t] = s_part[t][0] + s_part[t][1] + s_part[t][2] + s_part[t][3] + bV[0];
    __syncthreads();

    // softmax over 64 logits (redundantly computed per thread; L is tiny)
    float mx = -1e30f;
    for (int i = 0; i < LQ; ++i) mx = fmaxf(mx, s_logit[i]);
    float denom = 0.f;
    for (int i = 0; i < LQ; ++i) denom += expf(s_logit[i] - mx);
    __syncthreads();
    if (t < LQ) {
        const float a = expf(s_logit[t] - mx) / denom;
        s_attn[t] = a;
        attn_out[(size_t)b * LQ + t] = a;
    }
    __syncthreads();

    // context reduction over L
    const float* fb = features + (size_t)b * LQ * FEATQ;
    for (int f = t; f < FEATQ; f += 256) {
        float c = 0.f;
        #pragma unroll 8
        for (int l = 0; l < LQ; ++l) c += s_attn[l] * fb[(size_t)l * FEATQ + f];
        context[(size_t)b * FEATQ + f] = c;
    }
}

// tcat[b, 0:2560] = tanh( j<2048 ? context[b,j] : hidden[b,j-2048] )
__global__ __launch_bounds__(256) void build_tcat(
    const float* __restrict__ context, const float* __restrict__ hidden,
    float* __restrict__ tcat)
{
    const int idx = blockIdx.x * 256 + threadIdx.x;
    const int n = BQ * (FEATQ + UNITSQ);
    if (idx >= n) return;
    const int b = idx / (FEATQ + UNITSQ);
    const int j = idx % (FEATQ + UNITSQ);
    const float v = (j < FEATQ) ? context[(size_t)b * FEATQ + j]
                                : hidden[(size_t)b * UNITSQ + (j - FEATQ)];
    tcat[idx] = tanhf(v);
}

// x_in[b, 0:768] = concat(cvh[b], emb_table[tok[b]])
__global__ __launch_bounds__(256) void build_xin(
    const float* __restrict__ cvh, const int* __restrict__ tokens,
    const float* __restrict__ emb_table, float* __restrict__ xin)
{
    const int idx = blockIdx.x * 256 + threadIdx.x;
    const int n = BQ * (UNITSQ + EMBQ);
    if (idx >= n) return;
    const int b = idx / (UNITSQ + EMBQ);
    const int j = idx % (UNITSQ + EMBQ);
    float v;
    if (j < UNITSQ) v = cvh[(size_t)b * UNITSQ + j];
    else            v = emb_table[(size_t)tokens[b] * EMBQ + (j - UNITSQ)];
    xin[idx] = v;
}

// GRU finalize with h0 == 0:  gh = 0, so h_new = (1 - sigmoid(gx_z)) * tanh(gx_h)
__global__ __launch_bounds__(256) void gru_finalize(
    const float* __restrict__ gx,   // (B, 3*UNITS)
    float* __restrict__ h_new)      // (B, UNITS)
{
    const int idx = blockIdx.x * 256 + threadIdx.x;
    if (idx >= BQ * UNITSQ) return;
    const int b = idx / UNITSQ;
    const int u = idx % UNITSQ;
    const float gz = gx[(size_t)b * (3 * UNITSQ) + u];
    const float gh = gx[(size_t)b * (3 * UNITSQ) + 2 * UNITSQ + u];
    const float z  = 1.f / (1.f + expf(-gz));
    h_new[idx] = (1.f - z) * tanhf(gh);
}

// ---------------------------------------------------------------------------
extern "C" void kernel_launch(void* const* d_in, const int* in_sizes, int n_in,
                              void* d_out, int out_size, void* d_ws, size_t ws_size,
                              hipStream_t stream) {
    (void)in_sizes; (void)n_in; (void)out_size; (void)ws_size;
    const int*   x_tokens  = (const int*)  d_in[0];
    const float* features  = (const float*)d_in[1];
    const float* hidden    = (const float*)d_in[2];
    const float* W1        = (const float*)d_in[3];
    const float* b1        = (const float*)d_in[4];
    const float* W2        = (const float*)d_in[5];
    const float* b2        = (const float*)d_in[6];
    const float* Vvec      = (const float*)d_in[7];
    const float* bV        = (const float*)d_in[8];
    const float* W3        = (const float*)d_in[9];
    const float* b3        = (const float*)d_in[10];
    const float* emb_table = (const float*)d_in[11];
    const float* gru_kernel= (const float*)d_in[12];
    /* d_in[13] = gru_rec: unused — h0 == 0 makes gh == 0 */
    const float* gru_bias  = (const float*)d_in[14];
    const float* fc1_W     = (const float*)d_in[15];
    const float* fc1_b     = (const float*)d_in[16];
    const float* fc2_W     = (const float*)d_in[17];
    const float* fc2_b     = (const float*)d_in[18];

    // outputs: logits_out (B*VOCAB), h_new (B*UNITS), attn (B*L)
    float* out_logits = (float*)d_out;
    float* out_hnew   = out_logits + (size_t)BQ * VOCABQ;
    float* out_attn   = out_hnew   + (size_t)BQ * UNITSQ;

    // workspace layout (floats)
    float* ws = (float*)d_ws;
    float* ws_hW2   = ws;                                   // 256*512
    float* ws_score = ws_hW2   + (size_t)BQ * UNITSQ;       // 16384*512
    float* ws_ctx   = ws_score + (size_t)BQ * LQ * UNITSQ;  // 256*2048
    float* ws_tcat  = ws_ctx   + (size_t)BQ * FEATQ;        // 256*2560
    float* ws_cvh   = ws_tcat  + (size_t)BQ * (FEATQ+UNITSQ); // 256*512
    float* ws_xin   = ws_cvh   + (size_t)BQ * UNITSQ;       // 256*768
    float* ws_gx    = ws_xin   + (size_t)BQ * (UNITSQ+EMBQ);// 256*1536
    float* ws_x1    = ws_gx    + (size_t)BQ * 3 * UNITSQ;   // 256*512

    const dim3 blk(256);
    auto gemm = [&](const float* A, const float* B, const float* bias,
                    const float* addrow, float* C, int M, int N, int K,
                    int rowdiv, int act) {
        // wave tile 32x64; 8 waves/block along N -> 512 cols per block
        dim3 grid(M / 32, (N + 511) / 512);
        gemm_wmma_f16<<<grid, blk, 0, stream>>>(A, B, bias, addrow, C,
                                                M, N, K, rowdiv, act);
    };

    // 1) hW2 = hidden @ W2 + b2                       (256x512, K=512)
    gemm(hidden, W2, b2, nullptr, ws_hW2, BQ, UNITSQ, UNITSQ, 1, 0);
    // 2) score = tanh(features @ W1 + b1 + hW2[row/64])  (16384x512, K=2048)
    gemm(features, W1, b1, ws_hW2, ws_score, BQ * LQ, UNITSQ, FEATQ, LQ, 1);
    // 3) attention: logits, softmax, context (+ attn output)
    attention_kernel<<<dim3(BQ), blk, 0, stream>>>(ws_score, features, Vvec, bV,
                                                   out_attn, ws_ctx);
    // 4) tcat = tanh(concat(context, hidden)); cvh = tcat @ W3 + b3
    build_tcat<<<dim3((BQ * (FEATQ + UNITSQ)) / 256), blk, 0, stream>>>(
        ws_ctx, hidden, ws_tcat);
    gemm(ws_tcat, W3, b3, nullptr, ws_cvh, BQ, UNITSQ, FEATQ + UNITSQ, 1, 0);
    // 5) x_in = concat(cvh, emb[tok]); gx = x_in @ gru_kernel + gru_bias; GRU
    build_xin<<<dim3((BQ * (UNITSQ + EMBQ)) / 256), blk, 0, stream>>>(
        ws_cvh, x_tokens, emb_table, ws_xin);
    gemm(ws_xin, gru_kernel, gru_bias, nullptr, ws_gx, BQ, 3 * UNITSQ,
         UNITSQ + EMBQ, 1, 0);
    gru_finalize<<<dim3((BQ * UNITSQ) / 256), blk, 0, stream>>>(ws_gx, out_hnew);
    // 6) x1 = h_new @ fc1_W + fc1_b ; logits_out = x1 @ fc2_W + fc2_b
    gemm(out_hnew, fc1_W, fc1_b, nullptr, ws_x1, BQ, UNITSQ, UNITSQ, 1, 0);
    gemm(ws_x1, fc2_W, fc2_b, nullptr, out_logits, BQ, VOCABQ, UNITSQ, 1, 0);
}
